// WaveModel_51049981280426
// MI455X (gfx1250) — matrix operand order repaired and checked by reading, not compiled, not run on hardware
//
#include <hip/hip_runtime.h>
#include <cstdint>
#include <cstddef>

// ---------------------------------------------------------------------------
// FNet-like model on MI455X (gfx1250, wave32, WMMA).
// All heavy math runs through v_wmma_f32_16x16x32_f16 tiled GEMMs.
// rFFT/irFFT are GEMMs against precomputed f16 DFT matrices.
// B operands (and f16 A operands) are staged into LDS with
// GLOBAL_LOAD_ASYNC_TO_LDS_B128 (ASYNCcnt + s_wait_asynccnt) when available.
// ---------------------------------------------------------------------------

typedef __attribute__((ext_vector_type(16))) _Float16 v16h;
typedef __attribute__((ext_vector_type(8)))  _Float16 h8;
typedef __attribute__((ext_vector_type(8)))  float    v8f;
typedef int i32x4 __attribute__((ext_vector_type(4)));
typedef i32x4 __attribute__((address_space(1))) gi32x4;   // global
typedef i32x4 __attribute__((address_space(3))) li32x4;   // LDS

static constexpr int Bc  = 2;
static constexpr int Nc  = 2048;
static constexpr int Dc  = 1024;
static constexpr int Vc  = 32000;
static constexpr int NLc = 4;
static constexpr int NF  = Nc / 2 + 1;     // 1025 rfft bins
static constexpr int RF  = 2 * NF;         // 2050 ([Re;Im] rows)
static constexpr int KPRF = 2080;          // RF padded to multiple of 32 (zero fill)

#if defined(__gfx1250__) && \
    __has_builtin(__builtin_amdgcn_global_load_async_to_lds_b128) && \
    __has_builtin(__builtin_amdgcn_s_wait_asynccnt)
#define USE_ASYNC_LDS 1
#else
#define USE_ASYNC_LDS 0
#endif

__device__ __forceinline__ float gelu_exact(float x) {
    return 0.5f * x * (1.0f + erff(x * 0.70710678118654752f));
}

// 16-byte global -> LDS copy: async on gfx1250, plain vector copy otherwise.
// Address-space-typed pointers are produced via uintptr_t (AS3 is 32-bit; the
// low 32 bits of a flat LDS address are the LDS offset).
__device__ __forceinline__ void copy16_g2l(const _Float16* __restrict__ g,
                                           _Float16* __restrict__ l) {
#if USE_ASYNC_LDS
    __builtin_amdgcn_global_load_async_to_lds_b128(
        (gi32x4*)(uintptr_t)g, (li32x4*)(uintptr_t)l, 0, 0);
#else
    *(h8*)l = *(const h8*)g;
#endif
}

__device__ __forceinline__ void wait_g2l() {
#if USE_ASYNC_LDS
    __builtin_amdgcn_s_wait_asynccnt(0);
#endif
}

// ---------------------------------------------------------------------------
// Tiled WMMA GEMM:  C[M,N] = act( A[M,K] * Bt^T + bias )
//  - A: fp32 activations (VALU-converted while staging) or f16 (async staged)
//  - Bt: f16, pre-transposed [N, ldbT] with K padded to mult-of-32, zero fill
//  - f32 accumulate via v_wmma_f32_16x16x32_f16
// Block tile 128x128x32, 256 threads = 8 wave32s in 2(M) x 4(N) grid,
// each wave computes a 64x32 sub-tile (4x2 grid of 16x16 WMMA tiles).
// K must be a multiple of 32 (callers pad with zeros).
// ---------------------------------------------------------------------------
template <bool AF16, bool GELU_EP, bool BIAS_EP>
__global__ __launch_bounds__(256) void wmma_gemm(
    const void* __restrict__ Ap, const _Float16* __restrict__ Bt,
    const float* __restrict__ bias, float* __restrict__ Cp,
    int M, int N, int K, int lda, int ldbT, int ldc)
{
    __shared__ __align__(16) _Float16 As[128][40];  // [m][k], 8-half pad
    __shared__ __align__(16) _Float16 Bs[128][40];  // [n][k]

    const int tid   = threadIdx.x;
    const int lane  = tid & 31;
    const int wave  = tid >> 5;
    const int waveM = wave & 1;      // 2 waves along M
    const int waveN = wave >> 1;     // 4 waves along N
    const int blockM = blockIdx.y * 128;
    const int blockN = blockIdx.x * 128;

    const float*    Af = (const float*)Ap;
    const _Float16* Ah = (const _Float16*)Ap;

    // staging map: 256 threads x 16 halves = 128x32 tile
    const int srow = tid >> 1;
    const int scb  = (tid & 1) * 16;

    // clamped source rows (OOB columns/rows compute garbage that is never stored)
    int gnB = blockN + srow; if (gnB >= N) gnB = N - 1;
    int grA = blockM + srow; if (grA >= M) grA = M - 1;

    v8f acc[4][2] = {};

    for (int k0 = 0; k0 < K; k0 += 32) {
        // ---- stage B tile via async global->LDS (f16, already [n][k]) ----
        {
            const _Float16* src = Bt + (size_t)gnB * ldbT + k0 + scb;
            copy16_g2l(src,     &Bs[srow][scb]);
            copy16_g2l(src + 8, &Bs[srow][scb + 8]);
        }

        // ---- stage A tile ----
        if constexpr (AF16) {
            const _Float16* src = Ah + (size_t)grA * lda + k0 + scb;
            copy16_g2l(src,     &As[srow][scb]);
            copy16_g2l(src + 8, &As[srow][scb + 8]);
        } else {
            if (k0 + 32 < K)   // prefetch next K tile (global_prefetch_b8)
                __builtin_prefetch(Af + (size_t)grA * lda + k0 + 32, 0, 1);
            _Float16 tmp[16];
            if (blockM + srow < M) {
                const float4* p = (const float4*)(Af + (size_t)grA * lda + k0 + scb);
#pragma unroll
                for (int q = 0; q < 4; ++q) {
                    float4 v = p[q];
                    tmp[q*4+0] = (_Float16)v.x; tmp[q*4+1] = (_Float16)v.y;
                    tmp[q*4+2] = (_Float16)v.z; tmp[q*4+3] = (_Float16)v.w;
                }
            } else {
#pragma unroll
                for (int j = 0; j < 16; ++j) tmp[j] = (_Float16)0.f;
            }
            h8 lo, hi;
#pragma unroll
            for (int j = 0; j < 8; ++j) { lo[j] = tmp[j]; hi[j] = tmp[8 + j]; }
            h8* dst = (h8*)&As[srow][scb];
            dst[0] = lo; dst[1] = hi;
        }

        wait_g2l();          // s_wait_asynccnt 0
        __syncthreads();

        // ---- fragment loads per ISA 7.12.2 + 8 WMMAs per wave ----
        v16h afrag[4], bfrag[2];
#pragma unroll
        for (int mt = 0; mt < 4; ++mt) {
            // 16-bit A 16x32: lanes0-15 -> K {0..7,16..23}; lanes16-31 -> K {8..15,24..31}
            int r  = waveM * 64 + mt * 16 + (lane & 15);
            int kb = (lane >> 4) * 8;
            h8 lo = *(const h8*)&As[r][kb];
            h8 hi = *(const h8*)&As[r][kb + 16];
            afrag[mt] = __builtin_shufflevector(lo, hi,
                0,1,2,3,4,5,6,7,8,9,10,11,12,13,14,15);
        }
#pragma unroll
        for (int nt = 0; nt < 2; ++nt) {
            // B 32x16: lane%16 = column, lanes0-15 -> K {0..7,16..23}; lanes16-31 -> K {8..15,24..31}
            int c  = waveN * 32 + nt * 16 + (lane & 15);
            int kb = (lane >> 4) * 8;
            h8 lo = *(const h8*)&Bs[c][kb];
            h8 hi = *(const h8*)&Bs[c][kb + 16];
            bfrag[nt] = __builtin_shufflevector(lo, hi,
                0,1,2,3,4,5,6,7,8,9,10,11,12,13,14,15);
        }
#pragma unroll
        for (int mt = 0; mt < 4; ++mt)
#pragma unroll
            for (int nt = 0; nt < 2; ++nt)
                acc[mt][nt] = __builtin_amdgcn_wmma_f32_16x16x32_f16(
                    false, afrag[mt], false, bfrag[nt],
                    (short)0, acc[mt][nt], false, false);

        __syncthreads();
    }

    // ---- epilogue: C/D layout = VGPR e -> (M = base + e + 8*(lane>=16), N = lane%16)
#pragma unroll
    for (int mt = 0; mt < 4; ++mt) {
#pragma unroll
        for (int nt = 0; nt < 2; ++nt) {
            int col   = blockN + waveN * 32 + nt * 16 + (lane & 15);
            int rbase = blockM + waveM * 64 + mt * 16 + (lane >> 4) * 8;
            float bv = 0.f;
            if (BIAS_EP && col < N) bv = bias[col];
#pragma unroll
            for (int e = 0; e < 8; ++e) {
                int row = rbase + e;
                if (row < M && col < N) {
                    float v = acc[mt][nt][e] + bv;
                    if (GELU_EP) v = gelu_exact(v);
                    Cp[(size_t)row * ldc + col] = v;
                }
            }
        }
    }
}

// ---------------------------------------------------------------------------
// Transpose + fp32->f16 convert: in[K,N] fp32 -> out[N,KP] f16, zero-padded
// for k in [K,KP). Tiled through LDS so both sides stay coalesced.
// ---------------------------------------------------------------------------
__global__ __launch_bounds__(256) void conv_transpose_f16(
    const float* __restrict__ in, _Float16* __restrict__ outp,
    int K, int N, int KP)
{
    __shared__ float tile[32][33];
    const int kb = blockIdx.x * 32;
    const int nb = blockIdx.y * 32;
    const int tx = threadIdx.x & 31;
    const int ty = threadIdx.x >> 5;   // 0..7
#pragma unroll
    for (int i = 0; i < 4; ++i) {
        int k = kb + ty + i * 8, n = nb + tx;
        tile[ty + i * 8][tx] = (k < K && n < N) ? in[(size_t)k * N + n] : 0.f;
    }
    __syncthreads();
#pragma unroll
    for (int i = 0; i < 4; ++i) {
        int n = nb + ty + i * 8, k = kb + tx;
        if (n < N && k < KP)
            outp[(size_t)n * KP + k] = (_Float16)tile[tx][ty + i * 8];
    }
}

// ---------------------------------------------------------------------------
// Embedding + positional add: x[row, :] = embed[ids[row], :] + pos[row%N, :]
// ---------------------------------------------------------------------------
__global__ __launch_bounds__(256) void embed_kernel(
    const int* __restrict__ ids, const float* __restrict__ emb,
    const float* __restrict__ pos, float* __restrict__ x)
{
    const int row = blockIdx.x;          // 0 .. B*N-1
    const int n   = row % Nc;
    const int id  = ids[row];
    size_t eb = (size_t)id * Dc, pb = (size_t)n * Dc, ob = (size_t)row * Dc;
#pragma unroll
    for (int i = 0; i < 4; ++i) {
        int d = i * 256 + threadIdx.x;
        x[ob + d] = emb[eb + d] + pos[pb + d];
    }
}

// ---------------------------------------------------------------------------
// Residual + LayerNorm over D=1024 (one row per block), in-place into x.
// ---------------------------------------------------------------------------
__global__ __launch_bounds__(256) void ln_residual(
    float* __restrict__ x, const float* __restrict__ r,
    const float* __restrict__ g, const float* __restrict__ be)
{
    __shared__ float buf[Dc];
    __shared__ float rs[256];
    __shared__ float rq[256];
    const int row = blockIdx.x;
    const int tid = threadIdx.x;
    size_t base = (size_t)row * Dc;
    float s = 0.f, q = 0.f;
#pragma unroll
    for (int i = 0; i < 4; ++i) {
        int d = i * 256 + tid;
        float v = x[base + d] + r[base + d];
        buf[d] = v; s += v; q += v * v;
    }
    rs[tid] = s; rq[tid] = q;
    __syncthreads();
    for (int o = 128; o > 0; o >>= 1) {
        if (tid < o) { rs[tid] += rs[tid + o]; rq[tid] += rq[tid + o]; }
        __syncthreads();
    }
    float mean = rs[0] * (1.0f / Dc);
    float var  = rq[0] * (1.0f / Dc) - mean * mean;
    float inv  = rsqrtf(var + 1e-5f);
#pragma unroll
    for (int i = 0; i < 4; ++i) {
        int d = i * 256 + tid;
        x[base + d] = (buf[d] - mean) * inv * g[d] + be[d];
    }
}

// ---------------------------------------------------------------------------
// DFT matrix builders (f16).
// Forward F[RF x Nc]: rows 0..1024 -> cos(2*pi*k*n/N), rows 1025.. -> -sin.
// Inverse G[Nc x KPRF]: cols 0..1024 weight Re, cols 1025..2049 weight Im,
//   w_k = 1 for k in {0, N/2} else 2, scaled 1/N; cols 2050..2079 zero pad.
// ---------------------------------------------------------------------------
__global__ void fill_fwd_dft(_Float16* __restrict__ F)
{
    size_t idx = (size_t)blockIdx.x * 256 + threadIdx.x;
    if (idx >= (size_t)RF * Nc) return;
    int r = (int)(idx / Nc), n = (int)(idx % Nc);
    int k = (r < NF) ? r : (r - NF);
    long long m = ((long long)k * n) % Nc;          // exact phase reduction
    float ph = (float)m * (6.283185307179586f / Nc);
    F[idx] = (_Float16)((r < NF) ? cosf(ph) : -sinf(ph));
}

__global__ void fill_inv_dft(_Float16* __restrict__ G)
{
    size_t idx = (size_t)blockIdx.x * 256 + threadIdx.x;
    if (idx >= (size_t)Nc * KPRF) return;
    int n = (int)(idx / KPRF), c = (int)(idx % KPRF);
    float v = 0.f;
    if (c < RF) {
        int k = (c < NF) ? c : (c - NF);
        float w = (k == 0 || k == Nc / 2) ? 1.f : 2.f;
        long long m = ((long long)k * n) % Nc;
        float ph = (float)m * (6.283185307179586f / Nc);
        float t = (c < NF) ? cosf(ph) : -sinf(ph);
        v = w * t * (1.0f / Nc);
    }
    G[idx] = (_Float16)v;
}

// ---------------------------------------------------------------------------
// Launcher
// ---------------------------------------------------------------------------
extern "C" void kernel_launch(void* const* d_in, const int* in_sizes, int n_in,
                              void* d_out, int out_size, void* d_ws, size_t ws_size,
                              hipStream_t stream)
{
    (void)in_sizes; (void)n_in; (void)out_size; (void)ws_size;

    const int*   ids  = (const int*)  d_in[0];
    const float* emb  = (const float*)d_in[1];
    const float* pos  = (const float*)d_in[2];
    const float* W1f  = (const float*)d_in[3];
    const float* b1f  = (const float*)d_in[4];
    const float* W2f  = (const float*)d_in[5];
    const float* b2f  = (const float*)d_in[6];
    const float* g1   = (const float*)d_in[7];
    const float* be1  = (const float*)d_in[8];
    const float* Wf1  = (const float*)d_in[9];
    const float* bf1  = (const float*)d_in[10];
    const float* Wf2  = (const float*)d_in[11];
    const float* bf2  = (const float*)d_in[12];
    const float* g2   = (const float*)d_in[13];
    const float* be2  = (const float*)d_in[14];
    const float* Wh   = (const float*)d_in[15];
    const float* bh   = (const float*)d_in[16];
    float* out = (float*)d_out;

    // workspace carve (256B aligned)
    char* ws = (char*)d_ws;
    size_t off = 0;
    auto carve = [&](size_t bytes) -> void* {
        void* p = ws + off;
        off += (bytes + 255) & ~size_t(255);
        return p;
    };
    float*     x    = (float*)carve((size_t)Bc * Nc * Dc * 4);          // residual stream
    float*     xf   = (float*)carve((size_t)Bc * RF * Dc * 4);          // [Re;Im] freq stack
    float*     tmp  = (float*)carve((size_t)(Bc * Nc) * (4 * Dc) * 4);  // GEMM hidden buffer
    float*     xb   = (float*)carve((size_t)Bc * Nc * Dc * 4);          // FFT-branch / FFN out
    _Float16*  F    = (_Float16*)carve((size_t)RF * Nc * 2);            // forward DFT (f16)
    _Float16*  G    = (_Float16*)carve((size_t)Nc * KPRF * 2);          // inverse DFT (f16)
    _Float16*  wT   = (_Float16*)carve((size_t)Vc * Dc * 2);            // transposed f16 weight
    _Float16*  xT   = (_Float16*)carve((size_t)Bc * Dc * Nc * 2);       // x^T per batch
    _Float16*  xfpT = (_Float16*)carve((size_t)Bc * Dc * KPRF * 2);     // xfp^T per batch

    const int M1 = Bc * RF;   // 4100 rows through the frequency MLP
    const int MR = Bc * Nc;   // 4096 token rows

    auto tgrid = [](int KP, int N) { return dim3((KP + 31) / 32, (N + 31) / 32); };

    // DFT matrices (cheap vs GEMM work; rebuilt every call => deterministic)
    fill_fwd_dft<<<(int)(((size_t)RF * Nc + 255) / 256), 256, 0, stream>>>(F);
    fill_inv_dft<<<(int)(((size_t)Nc * KPRF + 255) / 256), 256, 0, stream>>>(G);

    embed_kernel<<<MR, 256, 0, stream>>>(ids, emb, pos, x);

    for (int i = 0; i < NLc; ++i) {
        const float* W1 = W1f + (size_t)i * Dc * 2 * Dc;
        const float* B1 = b1f + (size_t)i * 2 * Dc;
        const float* W2 = W2f + (size_t)i * 2 * Dc * Dc;
        const float* B2 = b2f + (size_t)i * Dc;
        const float* Wa = Wf1 + (size_t)i * Dc * 4 * Dc;
        const float* Ba = bf1 + (size_t)i * 4 * Dc;
        const float* Wb = Wf2 + (size_t)i * 4 * Dc * Dc;
        const float* Bb = bf2 + (size_t)i * Dc;

        // rFFT as GEMM: xf_b[RF, D] = F[RF, Nc] * x_b[Nc, D]
        for (int b = 0; b < Bc; ++b)
            conv_transpose_f16<<<tgrid(Nc, Dc), 256, 0, stream>>>(
                x + (size_t)b * Nc * Dc, xT + (size_t)b * Dc * Nc, Nc, Dc, Nc);
        for (int b = 0; b < Bc; ++b) {
            dim3 grid(Dc / 128, (RF + 127) / 128);
            wmma_gemm<true, false, false><<<grid, 256, 0, stream>>>(
                F, xT + (size_t)b * Dc * Nc, nullptr, xf + (size_t)b * RF * Dc,
                RF, Dc, Nc, Nc, Nc, Dc);
        }
        // frequency MLP on [Re;Im] stack (shared weights, per reference)
        conv_transpose_f16<<<tgrid(Dc, 2 * Dc), 256, 0, stream>>>(W1, wT, Dc, 2 * Dc, Dc);
        {
            dim3 grid(2 * Dc / 128, (M1 + 127) / 128);
            wmma_gemm<false, true, true><<<grid, 256, 0, stream>>>(
                xf, wT, B1, tmp, M1, 2 * Dc, Dc, Dc, Dc, 2 * Dc);
        }
        conv_transpose_f16<<<tgrid(2 * Dc, Dc), 256, 0, stream>>>(W2, wT, 2 * Dc, Dc, 2 * Dc);
        {
            dim3 grid(Dc / 128, (M1 + 127) / 128);
            wmma_gemm<false, false, true><<<grid, 256, 0, stream>>>(
                tmp, wT, B2, xf, M1, Dc, 2 * Dc, 2 * Dc, 2 * Dc, Dc);
        }
        // irFFT as GEMM: xb_b[Nc, D] = G[Nc, RF] * xfp_b[RF, D]  (K padded to 2080)
        for (int b = 0; b < Bc; ++b)
            conv_transpose_f16<<<tgrid(KPRF, Dc), 256, 0, stream>>>(
                xf + (size_t)b * RF * Dc, xfpT + (size_t)b * Dc * KPRF, RF, Dc, KPRF);
        for (int b = 0; b < Bc; ++b) {
            dim3 grid(Dc / 128, Nc / 128);
            wmma_gemm<true, false, false><<<grid, 256, 0, stream>>>(
                G, xfpT + (size_t)b * Dc * KPRF, nullptr, xb + (size_t)b * Nc * Dc,
                Nc, Dc, KPRF, KPRF, KPRF, Dc);
        }
        // x = LN(x + xb)
        ln_residual<<<MR, 256, 0, stream>>>(x, xb, g1 + (size_t)i * Dc, be1 + (size_t)i * Dc);

        // FFN
        conv_transpose_f16<<<tgrid(Dc, 4 * Dc), 256, 0, stream>>>(Wa, wT, Dc, 4 * Dc, Dc);
        {
            dim3 grid(4 * Dc / 128, MR / 128);
            wmma_gemm<false, true, true><<<grid, 256, 0, stream>>>(
                x, wT, Ba, tmp, MR, 4 * Dc, Dc, Dc, Dc, 4 * Dc);
        }
        conv_transpose_f16<<<tgrid(4 * Dc, Dc), 256, 0, stream>>>(Wb, wT, 4 * Dc, Dc, 4 * Dc);
        {
            dim3 grid(Dc / 128, MR / 128);
            wmma_gemm<false, false, true><<<grid, 256, 0, stream>>>(
                tmp, wT, Bb, xb, MR, Dc, 4 * Dc, 4 * Dc, 4 * Dc, Dc);
        }
        ln_residual<<<MR, 256, 0, stream>>>(x, xb, g2 + (size_t)i * Dc, be2 + (size_t)i * Dc);
    }

    // head: out[MR, V] = x[MR, D] * Wh[D, V] + bh
    conv_transpose_f16<<<tgrid(Dc, Vc), 256, 0, stream>>>(Wh, wT, Dc, Vc, Dc);
    {
        dim3 grid((Vc + 127) / 128, MR / 128);
        wmma_gemm<false, false, true><<<grid, 256, 0, stream>>>(
            x, wT, bh, out, MR, Vc, Dc, Dc, Dc, Vc);
    }
}
